// MemoryEfficientAttention_58325655879859
// MI455X (gfx1250) — matrix operand order repaired and checked by reading, not compiled, not run on hardware
//
#include <hip/hip_runtime.h>
#include <hip/hip_fp16.h>
#include <stdint.h>

// ---------------- problem constants (from reference) ----------------
constexpr int BATCH = 4;
constexpr int SEQ   = 2048;
constexpr int DM    = 1024;
constexpr int NH    = 16;
constexpr int HD    = 64;     // DM / NH
constexpr int MROWS = BATCH * SEQ;  // 8192

typedef __attribute__((ext_vector_type(16))) _Float16 v16h;
typedef __attribute__((ext_vector_type(8)))  _Float16 v8h;
typedef __attribute__((ext_vector_type(8)))  float    v8f;

// ---------------- WMMA fragment helpers (wave32, 16x16x32 f16) ----------------
// A-matrix 16x32 (MxK), ISA 7.12.2: lanes 0-15 -> M=lane, K halves 0-7 / 16-23;
// lanes 16-31 -> same M, K halves 8-15 / 24-31. Ptr points at tile origin.
__device__ __forceinline__ v16h load_a_frag(const _Float16* __restrict__ A, int lda, int lane) {
  int m     = lane & 15;
  int khalf = (lane >> 4) << 3;  // 0 or 8
  const _Float16* p = A + m * lda + khalf;
  v8h lo = *(const v8h*)(p);        // K = khalf .. khalf+7
  v8h hi = *(const v8h*)(p + 16);   // K = 16+khalf .. 16+khalf+7
  v16h a;
#pragma unroll
  for (int i = 0; i < 8; ++i) { a[i] = lo[i]; a[i + 8] = hi[i]; }
  return a;
}

// B-matrix 32x16 (KxN) where B[k][n] = Wt[n][k]; Wt is row-major [N, K] and
// the pointer is at element (n0, k0). Lane owns column n = lane&15; lanes 0-15
// carry K 0-7 / 16-23, lanes 16-31 carry K 8-15 / 24-31 (same as A-fragment).
__device__ __forceinline__ v16h load_b_frag(const _Float16* __restrict__ Wt, int ldw, int lane) {
  int n  = lane & 15;
  int kh = (lane >> 4) << 3;  // 0 or 8
  const _Float16* p = Wt + n * ldw + kh;
  v8h lo = *(const v8h*)(p);
  v8h hi = *(const v8h*)(p + 16);
  v16h b;
#pragma unroll
  for (int i = 0; i < 8; ++i) { b[i] = lo[i]; b[i + 8] = hi[i]; }
  return b;
}

__device__ __forceinline__ v8f wmma16(v16h a, v16h b, v8f c) {
  return __builtin_amdgcn_wmma_f32_16x16x32_f16(false, a, false, b, (short)0, c, false, false);
}

// Async copy of one 128-bit chunk global -> LDS (gfx1250, tracked by ASYNCcnt).
// VDST carries the wave-relative LDS byte address; VADDR the 64-bit global addr.
__device__ __forceinline__ void async_b128(uint32_t lds_off, const void* gptr) {
  asm volatile("global_load_async_to_lds_b128 %0, %1, off"
               :: "v"(lds_off), "v"((unsigned long long)(uintptr_t)gptr)
               : "memory");
}

// ---------------- fp32 -> fp16 convert ----------------
__global__ __launch_bounds__(256) void cvt_f16_kernel(const float* __restrict__ src,
                                                      _Float16* __restrict__ dst, int n) {
  int i = blockIdx.x * 256 + threadIdx.x;
  if (i < n) dst[i] = (_Float16)src[i];
}

// ---------------- block-tiled WMMA GEMM: C = A * W^T + bias ----------------
// Block = 4 waves = 64(M) x 64(N) tile. The 64x32 fp16 weight tile is staged
// into LDS once per K-step via global_load_async_to_lds_b128 (double-buffered,
// overlapped with the WMMAs of the current step); each wave computes 16x64.
// MODE 0: fp16 out, head-split  [B,H,S,Hd]
// MODE 1: fp16 out, head-split transposed [B,H,Hd,S]   (for V)
// MODE 2: fp32 out, flat [M,N]                          (final projection)

// Stage the 64-row x 32-K fp16 weight tile (4 KB) for this block: 128 threads
// x 2 rounds x 16 B. Exactly 2 async instructions per wave per stage.
__device__ __forceinline__ void stage_w_tile(const _Float16* __restrict__ Wblk, int K,
                                             int k0, uint32_t lds_base, int t) {
#pragma unroll
  for (int rnd = 0; rnd < 2; ++rnd) {
    int row = rnd * 32 + (t >> 2);           // 0..63
    int seg = t & 3;                         // 16B segment of the 64B row
    uint32_t loff = lds_base + (uint32_t)(row * 64 + seg * 16);
    const _Float16* g = Wblk + (size_t)row * K + k0 + seg * 8;
    async_b128(loff, g);
  }
}

template <int MODE>
__global__ __launch_bounds__(128) void gemm16_kernel(const _Float16* __restrict__ A,
                                                     const _Float16* __restrict__ W,
                                                     const float* __restrict__ bias,
                                                     void* __restrict__ out,
                                                     int M, int N, int K) {
  __shared__ _Float16 Btile[2][64 * 32];
  int t    = threadIdx.x;
  int lane = t & 31;
  int wid  = t >> 5;
  int nblk = N >> 6;
  int tn   = blockIdx.x % nblk;
  int tm   = blockIdx.x / nblk;

  const _Float16* Arow = A + (size_t)(tm * 64 + wid * 16) * K;
  const _Float16* Wblk = W + (size_t)(tn * 64) * K;
  uint32_t lds0 = (uint32_t)(uintptr_t)(&Btile[0][0]);
  uint32_t lds1 = (uint32_t)(uintptr_t)(&Btile[1][0]);

  const int KSTEPS = K >> 5;
  stage_w_tile(Wblk, K, 0, lds0, t);

  v8f acc[4] = {{}, {}, {}, {}};
  for (int ks = 0; ks < KSTEPS; ++ks) {
    uint32_t nxt = (ks & 1) ? lds0 : lds1;
    if (ks + 1 < KSTEPS) {
      stage_w_tile(Wblk, K, (ks + 1) << 5, nxt, t);
      asm volatile("s_wait_asynccnt 0x2" ::: "memory");   // current buffer landed
    } else {
      asm volatile("s_wait_asynccnt 0x0" ::: "memory");
    }
    __syncthreads();   // cross-wave visibility of the staged tile

    const _Float16* Bl = &Btile[ks & 1][0];
    __builtin_prefetch(Arow + ((ks + 2) << 5), 0, 1);
    v16h a = load_a_frag(Arow + (ks << 5), K, lane);
#pragma unroll
    for (int ct = 0; ct < 4; ++ct)
      acc[ct] = wmma16(a, load_b_frag(Bl + ct * 16 * 32, 32, lane), acc[ct]);

    __syncthreads();   // all waves done reading before buffer is re-staged
  }

  int n    = lane & 15;
  int half = lane >> 4;
#pragma unroll
  for (int ct = 0; ct < 4; ++ct) {
    int col  = tn * 64 + ct * 16 + n;
    float bv = bias[col];
#pragma unroll
    for (int r = 0; r < 8; ++r) {
      int row = tm * 64 + wid * 16 + half * 8 + r;
      float v = acc[ct][r] + bv;
      if (MODE == 2) {
        ((float*)out)[(size_t)row * N + col] = v;
      } else {
        int b  = row / SEQ;
        int s  = row - b * SEQ;
        int h  = col >> 6;       // col / HD
        int hd = col & 63;       // col % HD
        if (MODE == 0)
          ((_Float16*)out)[(((size_t)b * NH + h) * SEQ + s) * HD + hd] = (_Float16)v;
        else
          ((_Float16*)out)[(((size_t)b * NH + h) * HD + hd) * SEQ + s] = (_Float16)v;
      }
    }
  }
}

// ---------------- fused flash attention ----------------
// One wave handles 16 query rows of one (b,h). Online softmax over 32-key
// blocks; probabilities bounce through LDS to become an A-fragment for P*V.
__global__ __launch_bounds__(128) void flash_attn_kernel(const _Float16* __restrict__ Qh,
                                                         const _Float16* __restrict__ Kh,
                                                         const _Float16* __restrict__ VhT,
                                                         _Float16* __restrict__ Aout) {
  __shared__ _Float16 plds[4][16 * 32];
  int lane = threadIdx.x & 31;
  int wid  = threadIdx.x >> 5;
  int gw   = blockIdx.x * 4 + wid;

  const int qtiles = SEQ / 16;        // 128
  int qt = gw % qtiles;
  int bh = gw / qtiles;               // 0 .. B*H-1
  int b  = bh / NH;
  int h  = bh - b * NH;

  const _Float16* Qb = Qh  + (size_t)bh * SEQ * HD + (size_t)(qt * 16) * HD;
  const _Float16* Kb = Kh  + (size_t)bh * SEQ * HD;
  const _Float16* Vb = VhT + (size_t)bh * HD * SEQ;
  _Float16* pt = plds[wid];

  // Q fragments for the whole Hd=64 (two K-steps), reused across all key blocks
  v16h qa0 = load_a_frag(Qb,      HD, lane);
  v16h qa1 = load_a_frag(Qb + 32, HD, lane);

  v8f o0 = {}, o1 = {}, o2 = {}, o3 = {};
  float m[8], l[8];
#pragma unroll
  for (int r = 0; r < 8; ++r) { m[r] = -3.0e38f; l[r] = 0.0f; }

  const float sc = 0.125f;  // 1/sqrt(64)

  for (int kb = 0; kb < SEQ; kb += 32) {
    // ---- scores: S = Q (16x64) * K^T (64x32), two 16x16 column tiles ----
    v8f s0 = {}, s1 = {};
    {
      v16h b00 = load_b_frag(Kb + (size_t)kb * HD,              HD, lane);
      v16h b01 = load_b_frag(Kb + (size_t)kb * HD + 32,         HD, lane);
      s0 = wmma16(qa0, b00, s0);
      s0 = wmma16(qa1, b01, s0);
      v16h b10 = load_b_frag(Kb + (size_t)(kb + 16) * HD,       HD, lane);
      v16h b11 = load_b_frag(Kb + (size_t)(kb + 16) * HD + 32,  HD, lane);
      s1 = wmma16(qa0, b10, s1);
      s1 = wmma16(qa1, b11, s1);
    }

    // ---- online softmax over this 32-key block ----
    float p0[8], p1[8], sca[8];
#pragma unroll
    for (int r = 0; r < 8; ++r) {
      float v0 = s0[r] * sc;
      float v1 = s1[r] * sc;
      float rm = fmaxf(v0, v1);
      rm = fmaxf(rm, __shfl_xor(rm, 1));
      rm = fmaxf(rm, __shfl_xor(rm, 2));
      rm = fmaxf(rm, __shfl_xor(rm, 4));
      rm = fmaxf(rm, __shfl_xor(rm, 8));   // row max over 16 lanes (one C-row group)
      float mn = fmaxf(m[r], rm);
      sca[r] = __expf(m[r] - mn);
      m[r]   = mn;
      p0[r] = __expf(v0 - mn);
      p1[r] = __expf(v1 - mn);
      float rs = p0[r] + p1[r];
      rs += __shfl_xor(rs, 1);
      rs += __shfl_xor(rs, 2);
      rs += __shfl_xor(rs, 4);
      rs += __shfl_xor(rs, 8);
      l[r] = l[r] * sca[r] + rs;
    }
#pragma unroll
    for (int r = 0; r < 8; ++r) {
      o0[r] *= sca[r]; o1[r] *= sca[r]; o2[r] *= sca[r]; o3[r] *= sca[r];
    }

    // ---- P (C-layout) -> LDS 16x32 row-major -> A-fragment ----
    int rb = (lane >> 4) * 8;
    int c  = lane & 15;
#pragma unroll
    for (int r = 0; r < 8; ++r) {
      pt[(rb + r) * 32 + c]      = (_Float16)p0[r];
      pt[(rb + r) * 32 + 16 + c] = (_Float16)p1[r];
    }
    asm volatile("s_wait_dscnt 0" ::: "memory");   // cross-lane LDS RAW inside the wave
    v16h pa = load_a_frag(pt, 32, lane);

    // ---- O += P (16x32) * V (32x64): four Hd column tiles ----
    o0 = wmma16(pa, load_b_frag(Vb + (size_t)0  * SEQ + kb, SEQ, lane), o0);
    o1 = wmma16(pa, load_b_frag(Vb + (size_t)16 * SEQ + kb, SEQ, lane), o1);
    o2 = wmma16(pa, load_b_frag(Vb + (size_t)32 * SEQ + kb, SEQ, lane), o2);
    o3 = wmma16(pa, load_b_frag(Vb + (size_t)48 * SEQ + kb, SEQ, lane), o3);
  }

  // ---- normalize and write to [B,S,D] fp16 for the output projection ----
  int n    = lane & 15;
  int half = lane >> 4;
#pragma unroll
  for (int r = 0; r < 8; ++r) {
    float inv = 1.0f / l[r];
    int row = qt * 16 + half * 8 + r;
    _Float16* dst = Aout + ((size_t)b * SEQ + row) * DM + h * HD;
    dst[0 * 16 + n] = (_Float16)(o0[r] * inv);
    dst[1 * 16 + n] = (_Float16)(o1[r] * inv);
    dst[2 * 16 + n] = (_Float16)(o2[r] * inv);
    dst[3 * 16 + n] = (_Float16)(o3[r] * inv);
  }
}

// ---------------- host launcher ----------------
extern "C" void kernel_launch(void* const* d_in, const int* in_sizes, int n_in,
                              void* d_out, int out_size, void* d_ws, size_t ws_size,
                              hipStream_t stream) {
  const float* q  = (const float*)d_in[0];
  const float* k  = (const float*)d_in[1];
  const float* v  = (const float*)d_in[2];
  const float* Wq = (const float*)d_in[3];
  const float* bq = (const float*)d_in[4];
  const float* Wk = (const float*)d_in[5];
  const float* bk = (const float*)d_in[6];
  const float* Wv = (const float*)d_in[7];
  const float* bv = (const float*)d_in[8];
  const float* Wo = (const float*)d_in[9];
  const float* bo = (const float*)d_in[10];

  const size_t nX = (size_t)MROWS * DM;   // 8 Mi elements
  const size_t nW = (size_t)DM * DM;      // 1 Mi elements

  _Float16* ws = (_Float16*)d_ws;
  _Float16* qf   = ws;               ws += nX;
  _Float16* kf   = ws;               ws += nX;
  _Float16* vf   = ws;               ws += nX;
  _Float16* Wq16 = ws;               ws += nW;
  _Float16* Wk16 = ws;               ws += nW;
  _Float16* Wv16 = ws;               ws += nW;
  _Float16* Wo16 = ws;               ws += nW;
  _Float16* Qh   = ws;               ws += nX;   // [B,H,S,Hd]
  _Float16* Kh   = ws;               ws += nX;   // [B,H,S,Hd]
  _Float16* VhT  = ws;               ws += nX;   // [B,H,Hd,S]
  _Float16* Aout = ws;               ws += nX;   // [B,S,D]

  // 1) convert inputs + weights to fp16
  {
    int blkX = (int)((nX + 255) / 256);
    int blkW = (int)((nW + 255) / 256);
    cvt_f16_kernel<<<blkX, 256, 0, stream>>>(q,  qf,   (int)nX);
    cvt_f16_kernel<<<blkX, 256, 0, stream>>>(k,  kf,   (int)nX);
    cvt_f16_kernel<<<blkX, 256, 0, stream>>>(v,  vf,   (int)nX);
    cvt_f16_kernel<<<blkW, 256, 0, stream>>>(Wq, Wq16, (int)nW);
    cvt_f16_kernel<<<blkW, 256, 0, stream>>>(Wk, Wk16, (int)nW);
    cvt_f16_kernel<<<blkW, 256, 0, stream>>>(Wv, Wv16, (int)nW);
    cvt_f16_kernel<<<blkW, 256, 0, stream>>>(Wo, Wo16, (int)nW);
  }

  // 2) Q/K/V projections (async-staged WMMA GEMM, head-split outputs)
  {
    int blocks = (MROWS / 64) * (DM / 64);   // 2048
    gemm16_kernel<0><<<blocks, 128, 0, stream>>>(qf, Wq16, bq, Qh,  MROWS, DM, DM);
    gemm16_kernel<0><<<blocks, 128, 0, stream>>>(kf, Wk16, bk, Kh,  MROWS, DM, DM);
    gemm16_kernel<1><<<blocks, 128, 0, stream>>>(vf, Wv16, bv, VhT, MROWS, DM, DM);
  }

  // 3) fused flash attention
  {
    int waves  = BATCH * NH * (SEQ / 16);    // 8192
    int blocks = waves / 4;                  // 2048
    flash_attn_kernel<<<blocks, 128, 0, stream>>>(Qh, Kh, VhT, Aout);
  }

  // 4) output projection -> fp32 d_out
  {
    int blocks = (MROWS / 64) * (DM / 64);
    gemm16_kernel<2><<<blocks, 128, 0, stream>>>(Aout, Wo16, bo, (float*)d_out,
                                                 MROWS, DM, DM);
  }
}
